// GPT2Encoder_31679678775663
// MI455X (gfx1250) — matrix operand order repaired and checked by reading, not compiled, not run on hardware
//
#include <hip/hip_runtime.h>
#include <cstdint>

// ---------------------------------------------------------------------------
// Types for CDNA5 WMMA (wave32): v_wmma_f32_16x16x32_bf16
// ---------------------------------------------------------------------------
typedef __bf16 bf16_t;
typedef __attribute__((ext_vector_type(16))) __bf16 v16bf;
typedef __attribute__((ext_vector_type(8)))  float  v8f;

static __device__ __forceinline__ uint32_t bfround(float f) {
  union { float f; uint32_t u; } x; x.f = f;
  return (x.u + 0x7FFFu + ((x.u >> 16) & 1u)) >> 16;   // round-to-nearest-even
}
// pack two fp32 -> packed bf16x2 (lo in [15:0], hi in [31:16])
static __device__ __forceinline__ uint32_t bfpack(float lo, float hi) {
  return bfround(lo) | (bfround(hi) << 16);
}

static __device__ __forceinline__ float gelu_tanh(float x) {
  const float c0 = 0.7978845608028654f;   // sqrt(2/pi)
  const float c1 = 0.044715f;
  float t = tanhf(c0 * (x + c1 * x * x * x));
  return 0.5f * x * (1.0f + t);
}

union Frag { uint4 u[2]; v16bf v; };

// ---------------------------------------------------------------------------
// WMMA bf16 GEMM:  out[M,N] = act(A[M,K] @ W[K,N] + bias[N]) (+ res[M,N])
// 256 threads = 8 wave32. Block tile 128x128, K-step 32.
// Waves arranged 4(M) x 2(N); each wave owns a 32x64 tile = 2x4 WMMAs
// per K-step (A frags reused across 4 N-tiles, B frags across 2 M-tiles).
// A/B staged through LDS in bf16; B stored transposed so every fragment is
// two contiguous ds_load_b128. LDS stores are packed (b128 for A, b32 for B).
// ACT: 0 = none, 1 = tanh-GELU
// ---------------------------------------------------------------------------
#define TM 128
#define TN 128
#define TK 32
#define LDP (TK + 8)   // padded LDS row (bf16 elems): 80B rows keep 16B align

template<int ACT>
__global__ __launch_bounds__(256)
void gemm_bf16_wmma(const float* __restrict__ A, const float* __restrict__ W,
                    const float* __restrict__ bias, const float* res,
                    float* out, int M, int N, int K)
{
  __shared__ __align__(16) bf16_t As[TM][LDP];
  __shared__ __align__(16) bf16_t Bt[TN][LDP];   // transposed: Bt[n][k]

  const int tid   = threadIdx.x;
  const int lane  = tid & 31;
  const int wave  = tid >> 5;
  const int mTile = blockIdx.y * TM;
  const int nTile = blockIdx.x * TN;

  const int mw = (wave >> 1) * 32;   // 0,32,64,96
  const int nw = (wave & 1) * 64;    // 0,64

  const int la  = lane & 15;
  const int hi  = lane >> 4;
  const int kbA = hi * 8;    // A frag: K in {kbA..kbA+7, kbA+16..kbA+23}
  const int kbB = hi * 8;    // B frag: K in {kbB..kbB+7, kbB+16..kbB+23}

  v8f acc[2][4];
  #pragma unroll
  for (int mi = 0; mi < 2; ++mi)
    #pragma unroll
    for (int ni = 0; ni < 4; ++ni) acc[mi][ni] = v8f{};

  for (int k0 = 0; k0 < K; k0 += TK) {
    // ---- prefetch next K-step tiles into L2 (global_prefetch_b8) ----
    if (k0 + TK < K) {
      const int pr = tid >> 1;                    // 0..127
      __builtin_prefetch(A + (size_t)(mTile + pr) * K + (k0 + TK) +
                             (tid & 1) * 16, 0, 0);
      __builtin_prefetch(W + (size_t)(k0 + TK + (tid >> 3)) * N + nTile +
                             (tid & 7) * 16, 0, 0);
    }

    // ---- A tile: 128x32 fp32 -> bf16, packed ds_store_b128 ----
    #pragma unroll
    for (int it = 0; it < 2; ++it) {
      const int g   = tid + it * 256;             // 0..511
      const int row = g >> 2;                     // 0..127
      const int col = (g & 3) * 8;                // 0,8,16,24
      const float4 f0 = *(const float4*)(A + (size_t)(mTile + row) * K + k0 + col);
      const float4 f1 = *(const float4*)(A + (size_t)(mTile + row) * K + k0 + col + 4);
      uint4 p;
      p.x = bfpack(f0.x, f0.y); p.y = bfpack(f0.z, f0.w);
      p.z = bfpack(f1.x, f1.y); p.w = bfpack(f1.z, f1.w);
      *(uint4*)&As[row][col] = p;
    }
    // ---- B tile: 32x128 fp32, transposed into Bt[n][k], b32 stores ----
    #pragma unroll
    for (int it = 0; it < 2; ++it) {
      const int g  = tid + it * 256;              // 0..511
      const int kk = (g >> 5) * 2;                // 0,2,..,30
      const int nn = (g & 31) * 4;                // 0,4,..,124
      const float4 r0 = *(const float4*)(W + (size_t)(k0 + kk)     * N + nTile + nn);
      const float4 r1 = *(const float4*)(W + (size_t)(k0 + kk + 1) * N + nTile + nn);
      *(uint32_t*)&Bt[nn + 0][kk] = bfpack(r0.x, r1.x);
      *(uint32_t*)&Bt[nn + 1][kk] = bfpack(r0.y, r1.y);
      *(uint32_t*)&Bt[nn + 2][kk] = bfpack(r0.z, r1.z);
      *(uint32_t*)&Bt[nn + 3][kk] = bfpack(r0.w, r1.w);
    }
    __syncthreads();

    // ---- fragments (all contiguous 16B LDS reads) ----
    Frag a[2], b[4];
    #pragma unroll
    for (int mi = 0; mi < 2; ++mi) {
      a[mi].u[0] = *(const uint4*)&As[mw + 16 * mi + la][kbA];
      a[mi].u[1] = *(const uint4*)&As[mw + 16 * mi + la][kbA + 16];
    }
    #pragma unroll
    for (int ni = 0; ni < 4; ++ni) {
      b[ni].u[0] = *(const uint4*)&Bt[nw + 16 * ni + la][kbB];
      b[ni].u[1] = *(const uint4*)&Bt[nw + 16 * ni + la][kbB + 16];
    }
    // ---- 8 WMMAs per K-step ----
    #pragma unroll
    for (int mi = 0; mi < 2; ++mi)
      #pragma unroll
      for (int ni = 0; ni < 4; ++ni)
        acc[mi][ni] = __builtin_amdgcn_wmma_f32_16x16x32_bf16(
            false, a[mi].v, false, b[ni].v, (short)0, acc[mi][ni],
            false, false);
    __syncthreads();
  }

  // ---- epilogue: C/D layout -> VGPR r holds M = r + 8*hi, N = lane&15 ----
  const int rb = hi * 8;
  #pragma unroll
  for (int mi = 0; mi < 2; ++mi) {
    #pragma unroll
    for (int ni = 0; ni < 4; ++ni) {
      const int col = nTile + nw + 16 * ni + la;
      const float bv = bias[col];
      #pragma unroll
      for (int r = 0; r < 8; ++r) {
        const int row = mTile + mw + 16 * mi + rb + r;
        float v = acc[mi][ni][r] + bv;
        if (ACT == 1) v = gelu_tanh(v);
        const size_t o = (size_t)row * N + col;
        if (res) v += res[o];
        out[o] = v;
      }
    }
  }
}

// ---------------------------------------------------------------------------
// LayerNorm: one block per row of [rows, 1024]; float4 accesses.
// ---------------------------------------------------------------------------
__global__ __launch_bounds__(256)
void layernorm_kernel(const float* __restrict__ in, const float* __restrict__ g,
                      const float* __restrict__ b, float* __restrict__ out)
{
  const int D = 1024;
  __shared__ float red[256];
  const int tid = threadIdx.x;
  const float* x = in + (size_t)blockIdx.x * D;
  float* y = out + (size_t)blockIdx.x * D;

  const float4 xv = *(const float4*)(x + tid * 4);
  float s = xv.x + xv.y + xv.z + xv.w;
  red[tid] = s; __syncthreads();
  #pragma unroll
  for (int off = 128; off > 0; off >>= 1) {
    if (tid < off) red[tid] += red[tid + off];
    __syncthreads();
  }
  const float mean = red[0] * (1.0f / D);
  __syncthreads();

  float vs = (xv.x - mean) * (xv.x - mean) + (xv.y - mean) * (xv.y - mean) +
             (xv.z - mean) * (xv.z - mean) + (xv.w - mean) * (xv.w - mean);
  red[tid] = vs; __syncthreads();
  #pragma unroll
  for (int off = 128; off > 0; off >>= 1) {
    if (tid < off) red[tid] += red[tid + off];
    __syncthreads();
  }
  const float rstd = rsqrtf(red[0] * (1.0f / D) + 1e-5f);

  const float4 gv = *(const float4*)(g + tid * 4);
  const float4 bv = *(const float4*)(b + tid * 4);
  float4 o;
  o.x = (xv.x - mean) * rstd * gv.x + bv.x;
  o.y = (xv.y - mean) * rstd * gv.y + bv.y;
  o.z = (xv.z - mean) * rstd * gv.z + bv.z;
  o.w = (xv.w - mean) * rstd * gv.w + bv.w;
  *(float4*)(y + tid * 4) = o;
}

// ---------------------------------------------------------------------------
// Embedding gather: h[row,:] = wte[ids[row],:]  (float4 copy)
// ---------------------------------------------------------------------------
__global__ __launch_bounds__(256)
void embed_kernel(const int* __restrict__ ids, const float* __restrict__ wte,
                  float* __restrict__ h)
{
  const int D = 1024;
  const int row = blockIdx.x;
  const int id = ids[row];
  const float4* src = (const float4*)(wte + (size_t)id * D);
  float4* dst = (float4*)(h + (size_t)row * D);
  dst[threadIdx.x] = src[threadIdx.x];
}

// ---------------------------------------------------------------------------
// Causal attention, online softmax. One wave32 per query row.
// qkv layout: [B*S, 3072] with q=[0:1024), k=[1024:2048), v=[2048:3072).
// Lane holds 2 contiguous head dims (float2); wave32 shfl_xor reduction.
// ---------------------------------------------------------------------------
__global__ __launch_bounds__(256)
void attention_kernel(const float* __restrict__ qkv, float* __restrict__ attn)
{
  const int S = 1024, Hn = 16;
  const int gw   = blockIdx.x * 8 + (threadIdx.x >> 5);  // global wave id
  const int lane = threadIdx.x & 31;
  const int s = gw % S;
  const int h = (gw / S) % Hn;
  const int b = gw / (S * Hn);
  const int d0 = lane * 2;

  const float2 qv = *(const float2*)(qkv + ((size_t)b * S + s) * 3072 + h * 64 + d0);

  float m = -1e30f, l = 0.f, a0 = 0.f, a1 = 0.f;
  const size_t base = (size_t)b * S * 3072 + 1024 + h * 64 + d0;
  for (int j = 0; j <= s; ++j) {
    const float2 kv = *(const float2*)(qkv + base + (size_t)j * 3072);
    float dot = qv.x * kv.x + qv.y * kv.y;
    #pragma unroll
    for (int off = 16; off > 0; off >>= 1)
      dot += __shfl_xor(dot, off, 32);
    const float sc = dot * 0.125f;            // 1/sqrt(64)
    const float mn = fmaxf(m, sc);
    const float scale = __expf(m - mn);
    const float p = __expf(sc - mn);
    const float2 vv = *(const float2*)(qkv + base + (size_t)j * 3072 + 1024);
    a0 = a0 * scale + p * vv.x;
    a1 = a1 * scale + p * vv.y;
    l = l * scale + p;
    m = mn;
  }
  const float rl = 1.0f / l;
  float2 o; o.x = a0 * rl; o.y = a1 * rl;
  *(float2*)(attn + ((size_t)b * S + s) * 1024 + h * 64 + d0) = o;
}

// ---------------------------------------------------------------------------
// Host orchestration
// ---------------------------------------------------------------------------
extern "C" void kernel_launch(void* const* d_in, const int* in_sizes, int n_in,
                              void* d_out, int out_size, void* d_ws, size_t ws_size,
                              hipStream_t stream)
{
  (void)in_sizes; (void)n_in; (void)out_size; (void)ws_size;

  const int*   ids   = (const int*)  d_in[0];
  const float* wte   = (const float*)d_in[1];
  const float* ln1w  = (const float*)d_in[2];
  const float* ln1b  = (const float*)d_in[3];
  const float* attw  = (const float*)d_in[4];
  const float* attb  = (const float*)d_in[5];
  const float* projw = (const float*)d_in[6];
  const float* projb = (const float*)d_in[7];
  const float* ln2w  = (const float*)d_in[8];
  const float* ln2b  = (const float*)d_in[9];
  const float* fcw   = (const float*)d_in[10];
  const float* fcb   = (const float*)d_in[11];
  const float* fc2w  = (const float*)d_in[12];
  const float* fc2b  = (const float*)d_in[13];
  const float* lnfw  = (const float*)d_in[14];
  const float* lnfb  = (const float*)d_in[15];
  const float* headw = (const float*)d_in[16];
  const float* headb = (const float*)d_in[17];
  float* out = (float*)d_out;

  const size_t BS = 2048;               // B*S
  float* hbuf    = (float*)d_ws;                    //  8 MB
  float* xbuf    = hbuf    + BS * 1024;             //  8 MB
  float* qkvbuf  = xbuf    + BS * 1024;             // 24 MB
  float* attnbuf = qkvbuf  + BS * 3072;             //  8 MB
  float* midbuf  = attnbuf + BS * 1024;             // 32 MB

  embed_kernel<<<2048, 256, 0, stream>>>(ids, wte, hbuf);

  for (int l = 0; l < 12; ++l) {
    layernorm_kernel<<<2048, 256, 0, stream>>>(
        hbuf, ln1w + (size_t)l * 1024, ln1b + (size_t)l * 1024, xbuf);

    gemm_bf16_wmma<0><<<dim3(3072 / TN, 2048 / TM), 256, 0, stream>>>(
        xbuf, attw + (size_t)l * 1024 * 3072, attb + (size_t)l * 3072,
        nullptr, qkvbuf, 2048, 3072, 1024);

    attention_kernel<<<4096, 256, 0, stream>>>(qkvbuf, attnbuf);

    gemm_bf16_wmma<0><<<dim3(1024 / TN, 2048 / TM), 256, 0, stream>>>(
        attnbuf, projw + (size_t)l * 1024 * 1024, projb + (size_t)l * 1024,
        hbuf, hbuf, 2048, 1024, 1024);          // residual 1 (in-place on h)

    layernorm_kernel<<<2048, 256, 0, stream>>>(
        hbuf, ln2w + (size_t)l * 1024, ln2b + (size_t)l * 1024, xbuf);

    gemm_bf16_wmma<1><<<dim3(4096 / TN, 2048 / TM), 256, 0, stream>>>(
        xbuf, fcw + (size_t)l * 1024 * 4096, fcb + (size_t)l * 4096,
        nullptr, midbuf, 2048, 4096, 1024);     // fused tanh-GELU

    gemm_bf16_wmma<0><<<dim3(1024 / TN, 2048 / TM), 256, 0, stream>>>(
        midbuf, fc2w + (size_t)l * 4096 * 1024, fc2b + (size_t)l * 1024,
        hbuf, hbuf, 2048, 1024, 4096);          // residual 2 (in-place on h)
  }

  layernorm_kernel<<<2048, 256, 0, stream>>>(hbuf, lnfw, lnfb, xbuf);

  gemm_bf16_wmma<0><<<dim3(1024 / TN, 2048 / TM), 256, 0, stream>>>(
      xbuf, headw, headb, nullptr, out, 2048, 1024, 1024);
}
